// ImplicitRelationEncoder_17420387352924
// MI455X (gfx1250) — matrix operand order repaired and checked by reading, not compiled, not run on hardware
//
#include <hip/hip_runtime.h>

// ---------------------------------------------------------------- types
typedef __bf16 bf16;
typedef __bf16 v16bf __attribute__((ext_vector_type(16)));
typedef float  v8f   __attribute__((ext_vector_type(8)));
typedef int    v4i   __attribute__((vector_size(16)));

union Frag { uint4 q[2]; unsigned u[8]; v16bf v; };
union U4   { uint4 q; unsigned u[4]; };

#if __has_builtin(__builtin_amdgcn_global_load_async_to_lds_b128) && \
    __has_builtin(__builtin_amdgcn_s_wait_asynccnt)
#define USE_ASYNC_LDS 1
#else
#define USE_ASYNC_LDS 0
#endif

__device__ __forceinline__ unsigned rne_up(unsigned u) {
    return u + 0x7FFFu + ((u >> 16) & 1u);   // bf16 RNE in high half
}
// pack two f32 -> two bf16 (RNE): one v_perm_b32 merges the high halves
__device__ __forceinline__ unsigned pack2(float lo, float hi) {
    unsigned a = rne_up(__float_as_uint(lo));
    unsigned b = rne_up(__float_as_uint(hi));
    return __builtin_amdgcn_perm(b, a, 0x07060302u);  // {b.hi16, a.hi16}
}
__device__ __forceinline__ bf16 to_bf16(float x) {
    unsigned short h = (unsigned short)(rne_up(__float_as_uint(x)) >> 16);
    return __builtin_bit_cast(bf16, h);
}

// ---------------------------------------------------------------- problem dims
constexpr int B_ = 512, N_ = 36, VD_ = 2048, QD_ = 1024, D_ = 1024;
constexpr int H_ = 16, NG_ = 20, P_ = 64, DH_ = 64;
constexpr int M1 = B_ * N_;    // 18432
constexpr int MK = B_ * NG_;   // 10240

// ---------------------------------------------------------------- GEMM
// C(M,Nc) = A(M,K) * Bw(K,Nc). Block tile 128x128, K-step 32, 256 threads =
// 8 waves, wave tile 32x64 (2x4 grid of 16x16x32 bf16 WMMA).
// LDS layouts are fragment-ready: every WMMA fragment is two ds_load_b128;
// A tile is staged with GLOBAL_LOAD_ASYNC_TO_LDS_B128 (ASYNCcnt tracked).
constexpr int BM = 128, BN = 128, BK = 32;
constexpr int LDA_S = 40;  // A row stride in bf16 (80B, 16B aligned)
constexpr int LDBP  = 20;  // B col stride in dwords (80B, 16B aligned)

struct GemmEpilog {
    const float* bias;    // nullable: + bias[col]
    const float* qproj;   // nullable: + mask[row] * qproj[(row/nper)*Nc + col]
    const float* mask;
    int          nper;
    int          relu;
    float*       out_f;   // f32 output (ld = Nc)
    bf16*        out_b;   // nullable bf16 copy
};

template <bool A_F32>
__global__ __launch_bounds__(256) void gemm_bf16_kernel(
    const void* __restrict__ Ap, const bf16* __restrict__ Bw,
    int M, int Nc, int K, GemmEpilog ep)
{
    __shared__ __align__(16) bf16     As[2][BM * LDA_S];
    __shared__ __align__(16) unsigned Bp[2][BN * LDBP];

    const int tid  = threadIdx.x;
    const int bm   = blockIdx.x * BM;
    const int bn   = blockIdx.y * BN;
    const int lane = tid & 31;
    const int wid  = tid >> 5;
    const int wm   = (wid & 3) * 32;
    const int wn   = (wid >> 2) * 64;
    const int half = lane >> 4;
    const int l16  = lane & 15;

    // staging assignments
    const int arow = tid >> 1;           // 0..127
    const int acol = (tid & 1) * 16;     // 0 / 16
    const int bpr  = tid >> 4;           // 0..15  (K pair-row)
    const int bcg  = (tid & 15) * 8;     // col group (8 cols)

    v8f acc[2][4];
#pragma unroll
    for (int i = 0; i < 2; ++i)
#pragma unroll
        for (int j = 0; j < 4; ++j) acc[i][j] = (v8f)0.0f;

    auto stage = [&](int buf, int k0) {
        // ---- A tile: 128 x 32, row-major bf16, stride 40
        unsigned* ad = (unsigned*)&As[buf][arow * LDA_S + acol];
        if (A_F32) {
            const float* src = (const float*)Ap + (size_t)(bm + arow) * K + k0 + acol;
            float4 f0 = *(const float4*)(src + 0);
            float4 f1 = *(const float4*)(src + 4);
            float4 f2 = *(const float4*)(src + 8);
            float4 f3 = *(const float4*)(src + 12);
            U4 o0, o1;
            o0.u[0] = pack2(f0.x, f0.y); o0.u[1] = pack2(f0.z, f0.w);
            o0.u[2] = pack2(f1.x, f1.y); o0.u[3] = pack2(f1.z, f1.w);
            o1.u[0] = pack2(f2.x, f2.y); o1.u[1] = pack2(f2.z, f2.w);
            o1.u[2] = pack2(f3.x, f3.y); o1.u[3] = pack2(f3.z, f3.w);
            *(uint4*)(ad + 0) = o0.q;
            *(uint4*)(ad + 4) = o1.q;
        } else {
            const bf16* src = (const bf16*)Ap + (size_t)(bm + arow) * K + k0 + acol;
#if USE_ASYNC_LDS
            __builtin_amdgcn_global_load_async_to_lds_b128(
                (v4i*)(void*)src, (v4i*)(void*)(ad + 0), 0, 0);
            __builtin_amdgcn_global_load_async_to_lds_b128(
                (v4i*)(void*)(src + 8), (v4i*)(void*)(ad + 4), 0, 0);
#else
            uint4 u0 = *(const uint4*)src;
            uint4 u1 = *(const uint4*)(src + 8);
            *(uint4*)(ad + 0) = u0;
            *(uint4*)(ad + 4) = u1;
#endif
        }
        // ---- B tile: 32 x 128 -> packed transpose Bp[col][pair_row]
        // pair_row pr holds dword = (B[2pr][col], B[2pr+1][col])
        const bf16* bsrc = Bw + (size_t)(k0 + 2 * bpr) * Nc + bn + bcg;
        U4 ra, rb;
        ra.q = *(const uint4*)bsrc;
        rb.q = *(const uint4*)(bsrc + Nc);
        unsigned* bd = &Bp[buf][0];
#pragma unroll
        for (int j = 0; j < 4; ++j) {
            unsigned a = ra.u[j], b = rb.u[j];
            bd[(bcg + 2 * j + 0) * LDBP + bpr] = (a & 0xFFFFu) | (b << 16);
            bd[(bcg + 2 * j + 1) * LDBP + bpr] = (a >> 16) | (b & 0xFFFF0000u);
        }
    };

    stage(0, 0);
#if USE_ASYNC_LDS
    if (!A_F32) __builtin_amdgcn_s_wait_asynccnt(0);
#endif
    __syncthreads();

    const int nk = K / BK;
    for (int kk = 0; kk < nk; ++kk) {
        const int buf = kk & 1;
        if (kk + 1 < nk) stage(buf ^ 1, (kk + 1) * BK);

        // A fragments: two b128 per fragment (contiguous K: half h holds k 16h..16h+15)
        Frag afr[2];
#pragma unroll
        for (int tm = 0; tm < 2; ++tm) {
            const unsigned* ap =
                (const unsigned*)&As[buf][(wm + tm * 16 + l16) * LDA_S];
            afr[tm].q[0] = *(const uint4*)(ap + 8 * half);
            afr[tm].q[1] = *(const uint4*)(ap + 8 * half + 4);
        }
        // B fragments: two b128 per fragment from packed-transposed tile
        Frag bfr[4];
#pragma unroll
        for (int nt = 0; nt < 4; ++nt) {
            const unsigned* bp =
                &Bp[buf][(wn + nt * 16 + l16) * LDBP + 8 * half];
            bfr[nt].q[0] = *(const uint4*)(bp + 0);
            bfr[nt].q[1] = *(const uint4*)(bp + 4);
        }
#pragma unroll
        for (int tm = 0; tm < 2; ++tm)
#pragma unroll
            for (int nt = 0; nt < 4; ++nt)
                acc[tm][nt] = __builtin_amdgcn_wmma_f32_16x16x32_bf16(
                    false, afr[tm].v, false, bfr[nt].v,
                    (short)0, acc[tm][nt], false, false);
#if USE_ASYNC_LDS
        if (!A_F32) __builtin_amdgcn_s_wait_asynccnt(0);
#endif
        __syncthreads();
    }

    // epilogue: lane l16 = column, VGPR v = row v + 8*half
#pragma unroll
    for (int tm = 0; tm < 2; ++tm) {
#pragma unroll
        for (int nt = 0; nt < 4; ++nt) {
            const int col = bn + wn + nt * 16 + l16;
#pragma unroll
            for (int v = 0; v < 8; ++v) {
                const int row = bm + wm + tm * 16 + v + 8 * half;
                float x = acc[tm][nt][v];
                if (ep.bias)  x += ep.bias[col];
                if (ep.qproj) x += ep.mask[row] *
                                   ep.qproj[(size_t)(row / ep.nper) * Nc + col];
                if (ep.relu)  x = fmaxf(x, 0.0f);
                const size_t o = (size_t)row * Nc + col;
                if (ep.out_f) ep.out_f[o] = x;
                if (ep.out_b) ep.out_b[o] = to_bf16(x);
            }
        }
    }
}

// ---------------------------------------------------------------- helpers
__global__ void cast_f32_bf16_kernel(const float* __restrict__ src,
                                     bf16* __restrict__ dst) {
    const size_t i = ((size_t)blockIdx.x * 256 + threadIdx.x) * 4;
    float4 f = *(const float4*)(src + i);
    U4 o;
    o.u[0] = pack2(f.x, f.y);
    o.u[1] = pack2(f.z, f.w);
    *(uint2*)(dst + i) = make_uint2(o.u[0], o.u[1]);
}

// out_w (H,D,DH) f32 -> bf16 (D, H*DH): dst[f*1024 + h*64 + dh]
__global__ void perm_out_w_kernel(const float* __restrict__ w,
                                  bf16* __restrict__ dst) {
    const int idx = blockIdx.x * 256 + threadIdx.x;   // < 1048576
    const int dh = idx & 63;
    const int f  = (idx >> 6) & 1023;
    const int h  = idx >> 16;
    dst[(size_t)f * (H_ * DH_) + h * DH_ + dh] =
        to_bf16(w[(size_t)h * (D_ * DH_) + (size_t)f * DH_ + dh]);
}

__global__ void rowmask_kernel(const float* __restrict__ vis,
                               float* __restrict__ mask) {
    __shared__ float red[256];
    const int r = blockIdx.x;
    float s = 0.0f;
    for (int i = threadIdx.x; i < D_; i += 256) s += vis[(size_t)r * D_ + i];
    red[threadIdx.x] = s;
    __syncthreads();
    for (int w = 128; w > 0; w >>= 1) {
        if (threadIdx.x < w) red[threadIdx.x] += red[threadIdx.x + w];
        __syncthreads();
    }
    if (threadIdx.x == 0) mask[r] = (red[0] != 0.0f) ? 1.0f : 0.0f;
}

// pack ngf rows (b*36+m, m<20) -> contiguous (b*20+m)
__global__ void pack_ngf_kernel(const bf16* __restrict__ sf,
                                bf16* __restrict__ ngf) {
    const int r = blockIdx.x;               // 0..10239
    const int b = r / NG_, m = r % NG_;
    const bf16* s = sf + (size_t)(b * N_ + m) * D_;
    bf16* d = ngf + (size_t)r * D_;
    const int i = threadIdx.x * 4;
    *(uint2*)(d + i) = *(const uint2*)(s + i);
}

// ---------------------------------------------------------------- attention
__global__ __launch_bounds__(320) void attn_kernel(
    const float* __restrict__ qmat, const float* __restrict__ kmat,
    const float* __restrict__ kvmat, const float* __restrict__ sf,
    const float* __restrict__ visf, const float* __restrict__ pos_emb,
    const float* __restrict__ pos_w, const float* __restrict__ pos_b,
    const float* __restrict__ bias_w, const float* __restrict__ bias_b,
    const float* __restrict__ outb, float* __restrict__ outp)
{
    __shared__ float qs[D_];
    __shared__ float posw[P_ * H_];
    __shared__ float pose[NG_ * P_];
    __shared__ float aff[H_ * NG_];
    __shared__ float atw[H_ * NG_];

    const int bnrow = blockIdx.x;
    const int b = bnrow / N_;
    const int t = threadIdx.x;
    const size_t rowoff = (size_t)bnrow * D_;

    for (int i = t; i < D_; i += 320)        qs[i]   = qmat[rowoff + i];
    for (int i = t; i < P_ * H_; i += 320)   posw[i] = pos_w[i];
    for (int i = t; i < NG_ * P_; i += 320)  pose[i] = pos_emb[(size_t)bnrow * NG_ * P_ + i];
    __syncthreads();

    const float biasc = bias_w[0] + bias_b[0];

    if (t < H_ * NG_) {
        const int h = t / NG_, m = t % NG_;
        const float* krow = kmat + (size_t)(b * NG_ + m) * D_ + h * DH_;
        const float* qrow = qs + h * DH_;
        float qk = 0.0f;
#pragma unroll 8
        for (int d = 0; d < DH_; ++d) qk += qrow[d] * krow[d];
        qk *= 0.125f;
        float pf = pos_b[h];
        const float* pe = pose + m * P_;
#pragma unroll 8
        for (int p = 0; p < P_; ++p) pf += pe[p] * posw[p * H_ + h];
        pf = fmaxf(pf, 0.0f);
        aff[h * NG_ + m] = qk + __logf(fmaxf(pf, 1e-6f)) + biasc;
    }
    __syncthreads();

    if (t < H_) {
        float mx = -3.0e38f;
        for (int m = 0; m < NG_; ++m) mx = fmaxf(mx, aff[t * NG_ + m]);
        float s = 0.0f;
        for (int m = 0; m < NG_; ++m) {
            float e = __expf(aff[t * NG_ + m] - mx);
            atw[t * NG_ + m] = e;
            s += e;
        }
        const float r = 1.0f / s;
        for (int m = 0; m < NG_; ++m) atw[t * NG_ + m] *= r;
    }
    __syncthreads();

    for (int d = t; d < D_; d += 320) {
        const int h = d >> 6;
        float nb = outb[d];
        const float* kvp = kvmat + (size_t)b * NG_ * D_ + d;
#pragma unroll 5
        for (int m = 0; m < NG_; ++m) nb += atw[h * NG_ + m] * kvp[(size_t)m * D_];
        const float sfv = sf[rowoff + d];
        outp[rowoff + d] = visf[rowoff + d] + fmaxf(sfv + nb, 0.0f);
    }
}

// ---------------------------------------------------------------- launch
extern "C" void kernel_launch(void* const* d_in, const int* in_sizes, int n_in,
                              void* d_out, int out_size, void* d_ws, size_t ws_size,
                              hipStream_t stream) {
    (void)in_sizes; (void)n_in; (void)out_size; (void)ws_size;

    const float* visual  = (const float*)d_in[0];
    const float* pos_emb = (const float*)d_in[1];
    const float* question= (const float*)d_in[2];
    const float* v2out_w = (const float*)d_in[3];
    const float* v2out_b = (const float*)d_in[4];
    const float* self_w  = (const float*)d_in[5];
    const float* self_b  = (const float*)d_in[6];
    const float* q_w     = (const float*)d_in[7];
    const float* q_b     = (const float*)d_in[8];
    const float* k_w     = (const float*)d_in[9];
    const float* k_b     = (const float*)d_in[10];
    const float* pos_w   = (const float*)d_in[11];
    const float* pos_b   = (const float*)d_in[12];
    const float* bias_w  = (const float*)d_in[13];
    const float* bias_b  = (const float*)d_in[14];
    const float* out_w   = (const float*)d_in[15];
    const float* out_b   = (const float*)d_in[16];
    float* outp = (float*)d_out;

    char* base = (char*)d_ws;
    size_t off = 0;
    auto take = [&](size_t bytes) -> char* {
        char* p = base + off;
        off = (off + bytes + 255) & ~(size_t)255;
        return p;
    };
    bf16*  wv2   = (bf16*)take((size_t)VD_ * D_ * 2);
    bf16*  wself = (bf16*)take((size_t)(D_ + QD_) * D_ * 2);
    bf16*  wq    = (bf16*)take((size_t)D_ * D_ * 2);
    bf16*  wk    = (bf16*)take((size_t)D_ * D_ * 2);
    bf16*  wout  = (bf16*)take((size_t)D_ * H_ * DH_ * 2);
    bf16*  qst   = (bf16*)take((size_t)B_ * QD_ * 2);
    bf16*  visin = (bf16*)take((size_t)M1 * VD_ * 2);   // visual cast to bf16
    float* visf  = (float*)take((size_t)M1 * D_ * 4);
    bf16*  visb  = (bf16*)take((size_t)M1 * D_ * 2);
    float* maskb = (float*)take((size_t)M1 * 4);
    float* qproj = (float*)take((size_t)B_ * D_ * 4);
    float* sff   = (float*)take((size_t)M1 * D_ * 4);
    bf16*  sfb   = (bf16*)take((size_t)M1 * D_ * 2);
    float* qf    = (float*)take((size_t)M1 * D_ * 4);
    bf16*  ngfb  = (bf16*)take((size_t)MK * D_ * 2);
    float* kf    = (float*)take((size_t)MK * D_ * 4);
    float* kvf   = (float*)take((size_t)MK * D_ * 4);

    cast_f32_bf16_kernel<<<(VD_ * D_) / 1024, 256, 0, stream>>>(v2out_w, wv2);
    cast_f32_bf16_kernel<<<((D_ + QD_) * D_) / 1024, 256, 0, stream>>>(self_w, wself);
    cast_f32_bf16_kernel<<<(D_ * D_) / 1024, 256, 0, stream>>>(q_w, wq);
    cast_f32_bf16_kernel<<<(D_ * D_) / 1024, 256, 0, stream>>>(k_w, wk);
    cast_f32_bf16_kernel<<<(B_ * QD_) / 1024, 256, 0, stream>>>(question, qst);
    cast_f32_bf16_kernel<<<((size_t)M1 * VD_) / 1024, 256, 0, stream>>>(visual, visin);
    perm_out_w_kernel<<<(H_ * D_ * DH_) / 256, 256, 0, stream>>>(out_w, wout);

    // vis = relu(visual_bf16 @ v2out_w + b)
    {
        GemmEpilog ep{v2out_b, nullptr, nullptr, 1, 1, visf, visb};
        gemm_bf16_kernel<false><<<dim3(M1 / BM, D_ / BN), 256, 0, stream>>>(
            visin, wv2, M1, D_, VD_, ep);
    }
    rowmask_kernel<<<M1, 256, 0, stream>>>(visf, maskb);
    // qproj = question @ self_w[D:,:]
    {
        GemmEpilog ep{nullptr, nullptr, nullptr, 1, 0, qproj, nullptr};
        gemm_bf16_kernel<false><<<dim3(B_ / BM, D_ / BN), 256, 0, stream>>>(
            qst, wself + (size_t)D_ * D_, B_, D_, QD_, ep);
    }
    // self_feat = vis @ self_w[:D,:] + self_b + mask*qproj
    {
        GemmEpilog ep{self_b, qproj, maskb, N_, 0, sff, sfb};
        gemm_bf16_kernel<false><<<dim3(M1 / BM, D_ / BN), 256, 0, stream>>>(
            visb, wself, M1, D_, D_, ep);
    }
    // q = self_feat @ q_w + q_b
    {
        GemmEpilog ep{q_b, nullptr, nullptr, 1, 0, qf, nullptr};
        gemm_bf16_kernel<false><<<dim3(M1 / BM, D_ / BN), 256, 0, stream>>>(
            sfb, wq, M1, D_, D_, ep);
    }
    pack_ngf_kernel<<<MK, 256, 0, stream>>>(sfb, ngfb);
    // k = ngf @ k_w + k_b
    {
        GemmEpilog ep{k_b, nullptr, nullptr, 1, 0, kf, nullptr};
        gemm_bf16_kernel<false><<<dim3(MK / BM, D_ / BN), 256, 0, stream>>>(
            ngfb, wk, MK, D_, D_, ep);
    }
    // kv = ngf @ out_w_perm (fused attn@ngf@out_w reordering)
    {
        GemmEpilog ep{nullptr, nullptr, nullptr, 1, 0, kvf, nullptr};
        gemm_bf16_kernel<false><<<dim3(MK / BM, D_ / BN), 256, 0, stream>>>(
            ngfb, wout, MK, D_, D_, ep);
    }
    attn_kernel<<<M1, 320, 0, stream>>>(qf, kf, kvf, sff, visf, pos_emb,
                                        pos_w, pos_b, bias_w, bias_b, out_b, outp);
}
